// ScaledDotProductAttention_7129645711409
// MI455X (gfx1250) — matrix operand order repaired and checked by reading, not compiled, not run on hardware
//
#include <hip/hip_runtime.h>
#include <math.h>

// ===========================================================================
// Fused causal + key-padded attention, B=8, L=2048, D=256, fp32 in/out.
// Flash-attention single pass with online softmax.
//
// MI455X reasoning:
//  * 34.4 GFLOP vs ~67MB HBM -> compute-limited unless WMMA is used.
//  * f32 WMMA has K=4 per issue; f16 WMMA has K=32 -> stage Q/K/V/P as f16,
//    accumulate fp32, softmax in fp32 (inputs ~N(0,1)/16, f16-safe).
//  * wave32: 1 wave = 16 query rows; 8 waves/block = 128-row Q tile.
//  * K in LDS row-major f16; V in LDS TRANSPOSED so both GEMMs' B-fragments
//    are single contiguous 32B ds loads per lane.
//  * Row stats kept in WMMA C-layout; reductions = shfl_xor within 16-lane
//    halves (mask 1/2/4/8 never crosses the half boundary).
//  * Skipping fully-causal key tiles is exact: exp(-1e4) == 0.0f in fp32.
// ===========================================================================

typedef __attribute__((ext_vector_type(16))) _Float16 v16h;
typedef __attribute__((ext_vector_type(8)))  _Float16 v8h;
typedef __attribute__((ext_vector_type(8)))  float    v8f;
typedef __attribute__((ext_vector_type(4)))  float    v4f;

#define BATCH 8
#define SEQ   2048
#define DIM   256
#define NEGB  10000.0f
#define QTILE 128
#define KTILE 32
#define NWAVE 8

__launch_bounds__(256, 1)
__global__ void fa_fwd_kernel(const float* __restrict__ Qg,
                              const float* __restrict__ Kg,
                              const float* __restrict__ Vg,
                              const unsigned char* __restrict__ maskg, // jax bool = 1 byte
                              float* __restrict__ Og)
{
  __shared__ __align__(32) _Float16 Ksh[KTILE * DIM];        // [krow][d]
  __shared__ __align__(32) _Float16 Vsh[DIM * KTILE];        // [d][krow] (transposed)
  __shared__ __align__(32) _Float16 Psh[NWAVE * 16 * KTILE]; // per-wave P scratch

  const int tid  = threadIdx.x;
  const int wave = tid >> 5;
  const int lane = tid & 31;
  const int n    = lane & 15;   // N index / column-in-tile
  const int h    = lane >> 4;   // lane half
  const int b    = blockIdx.y;
  const int base = blockIdx.x * QTILE;
  const int q0   = base + wave * 16;  // this wave's first query row
  const int qhi  = q0 + 15;

  const float* Qb = Qg + (size_t)b * SEQ * DIM;
  const float* Kb = Kg + (size_t)b * SEQ * DIM;
  const float* Vb = Vg + (size_t)b * SEQ * DIM;
  const unsigned char* Mb = maskg + (size_t)b * SEQ;

  // ---- Q: 16x256 rows -> 8 A-fragments (16x32 f16).  A layout: lane m=n,
  // half h; element run is d = c*32 + [16h..16h+15]. ----
  v16h qf[8];
  {
    const float* qrow = Qb + (size_t)(q0 + n) * DIM;
#pragma unroll
    for (int c = 0; c < 8; ++c) {
      v4f a0 = *(const v4f*)(qrow + c * 32 + 16 * h);
      v4f a1 = *(const v4f*)(qrow + c * 32 + 16 * h + 4);
      v4f a2 = *(const v4f*)(qrow + c * 32 + 16 * h + 8);
      v4f a3 = *(const v4f*)(qrow + c * 32 + 16 * h + 12);
      v16h t;
#pragma unroll
      for (int i = 0; i < 4; ++i) {
        t[i]      = (_Float16)a0[i];
        t[4 + i]  = (_Float16)a1[i];
        t[8 + i]  = (_Float16)a2[i];
        t[12 + i] = (_Float16)a3[i];
      }
      qf[c] = t;
    }
  }

  v8f o[16];                       // O accumulators: 16 d-chunks of 16
#pragma unroll
  for (int t = 0; t < 16; ++t)
#pragma unroll
    for (int i = 0; i < 8; ++i) o[t][i] = 0.0f;

  float mrow[8], lrow[8];          // row stats, row = q0 + j + 8h
#pragma unroll
  for (int j = 0; j < 8; ++j) { mrow[j] = -INFINITY; lrow[j] = 0.0f; }

  const int ntiles = base / KTILE + QTILE / KTILE;  // keys 0 .. base+127
  const int ldr = tid >> 3;        // staging: row 0..31
  const int ldc = (tid & 7) * 32;  // staging: col start

  for (int kt = 0; kt < ntiles; ++kt) {
    const int kbase = kt * KTILE;
    __syncthreads();

    // ---- stage K tile, f32 -> f16, row-major ----
    {
      const float* src = Kb + (size_t)(kbase + ldr) * DIM + ldc;
      _Float16* dst = &Ksh[ldr * DIM + ldc];
#pragma unroll
      for (int i = 0; i < 4; ++i) {
        v4f f0 = *(const v4f*)(src + i * 8);
        v4f f1 = *(const v4f*)(src + i * 8 + 4);
        v8h hh;
#pragma unroll
        for (int k = 0; k < 4; ++k) { hh[k] = (_Float16)f0[k]; hh[4 + k] = (_Float16)f1[k]; }
        *(v8h*)(dst + i * 8) = hh;
      }
    }
    // ---- stage V tile, f32 -> f16, transposed [d][krow] ----
    {
      const float* src = Vb + (size_t)(kbase + ldr) * DIM + ldc;
#pragma unroll
      for (int i = 0; i < 8; ++i) {
        v4f f = *(const v4f*)(src + i * 4);
#pragma unroll
        for (int k = 0; k < 4; ++k)
          Vsh[(ldc + i * 4 + k) * KTILE + ldr] = (_Float16)f[k];
      }
    }
    if (kt + 1 < ntiles) {  // speculative prefetch of next tile
      __builtin_prefetch(Kb + (size_t)(kbase + KTILE + ldr) * DIM + ldc, 0, 0);
      __builtin_prefetch(Vb + (size_t)(kbase + KTILE + ldr) * DIM + ldc, 0, 0);
    }
    __syncthreads();

    if (kbase > qhi) continue;  // tile fully causal-masked for this wave (exact)

    // ---- S = Q K^T : two 16x16 f32 C tiles (cols n and 16+n of the tile).
    // B-frag for col tile ct: lane reads K[ct*16+n][c*32+16h .. +15] (32B). ----
    v8f s0, s1;
#pragma unroll
    for (int i = 0; i < 8; ++i) { s0[i] = 0.0f; s1[i] = 0.0f; }
#pragma unroll
    for (int c = 0; c < 8; ++c) {
      v16h bk0 = *(const v16h*)&Ksh[n * DIM + c * 32 + 16 * h];
      v16h bk1 = *(const v16h*)&Ksh[(16 + n) * DIM + c * 32 + 16 * h];
      s0 = __builtin_amdgcn_wmma_f32_16x16x32_f16(false, qf[c], false, bk0, (short)0, s0, false, false);
      s1 = __builtin_amdgcn_wmma_f32_16x16x32_f16(false, qf[c], false, bk1, (short)0, s1, false, false);
    }

    // ---- online softmax in C-layout; write P straight into LDS scratch ----
    const int col0 = kbase + n;
    const int col1 = kbase + 16 + n;
    const float bm0 = Mb[col0] ? 0.0f : -NEGB;
    const float bm1 = Mb[col1] ? 0.0f : -NEGB;
    _Float16* P = &Psh[wave * (16 * KTILE)];
#pragma unroll
    for (int j = 0; j < 8; ++j) {
      const int row = q0 + j + 8 * h;
      float e0 = s0[j] * 0.0625f + bm0 + (col0 > row ? -NEGB : 0.0f);
      float e1 = s1[j] * 0.0625f + bm1 + (col1 > row ? -NEGB : 0.0f);
      float rmax = fmaxf(e0, e1);
      rmax = fmaxf(rmax, __shfl_xor(rmax, 1, 32));
      rmax = fmaxf(rmax, __shfl_xor(rmax, 2, 32));
      rmax = fmaxf(rmax, __shfl_xor(rmax, 4, 32));
      rmax = fmaxf(rmax, __shfl_xor(rmax, 8, 32));
      const float mn  = fmaxf(mrow[j], rmax);
      const float scl = __expf(mrow[j] - mn);
      const float w0  = __expf(e0 - mn);
      const float w1  = __expf(e1 - mn);
      float rs = w0 + w1;
      rs += __shfl_xor(rs, 1, 32);
      rs += __shfl_xor(rs, 2, 32);
      rs += __shfl_xor(rs, 4, 32);
      rs += __shfl_xor(rs, 8, 32);
      lrow[j] = lrow[j] * scl + rs;
      mrow[j] = mn;
#pragma unroll
      for (int t = 0; t < 16; ++t) o[t][j] *= scl;   // rescale accumulators
      const int r = j + 8 * h;
      P[r * KTILE + n]      = (_Float16)w0;
      P[r * KTILE + 16 + n] = (_Float16)w1;
    }

    // Per-wave LDS round-trip: DS ops are in-order per wave, but make the
    // cross-lane visibility explicit with the CDNA5 split counter.
    asm volatile("s_wait_dscnt 0" ::: "memory");

    // ---- reload P as A-fragment (16x32): run [16h..16h+15] ----
    v16h pf;
    {
      v8h a0 = *(const v8h*)&P[n * KTILE + 16 * h];
      v8h a1 = *(const v8h*)&P[n * KTILE + 16 * h + 8];
#pragma unroll
      for (int i = 0; i < 8; ++i) { pf[i] = a0[i]; pf[8 + i] = a1[i]; }
    }

    // ---- O += P V : B-frag = Vsh[(t*16+n)*32 + 16h .. +15] (32B) ----
#pragma unroll
    for (int t = 0; t < 16; ++t) {
      v16h bv = *(const v16h*)&Vsh[(t * 16 + n) * KTILE + 16 * h];
      o[t] = __builtin_amdgcn_wmma_f32_16x16x32_f16(false, pf, false, bv, (short)0, o[t], false, false);
    }
  }

  // ---- epilogue: normalize by row sums, store fp32 ----
  float* Ob = Og + (size_t)b * SEQ * DIM;
#pragma unroll
  for (int j = 0; j < 8; ++j) {
    const float inv = 1.0f / lrow[j];
    float* orow = Ob + (size_t)(q0 + j + 8 * h) * DIM;
#pragma unroll
    for (int t = 0; t < 16; ++t)
      orow[t * 16 + n] = o[t][j] * inv;
  }
}

extern "C" void kernel_launch(void* const* d_in, const int* in_sizes, int n_in,
                              void* d_out, int out_size, void* d_ws, size_t ws_size,
                              hipStream_t stream) {
  (void)in_sizes; (void)n_in; (void)d_ws; (void)ws_size; (void)out_size;
  const float* Q = (const float*)d_in[0];
  const float* K = (const float*)d_in[1];
  const float* V = (const float*)d_in[2];
  const unsigned char* M = (const unsigned char*)d_in[3]; // bool mask, 1 byte/elem
  float* O = (float*)d_out;
  dim3 grid(SEQ / QTILE, BATCH);
  fa_fwd_kernel<<<grid, 256, 0, stream>>>(Q, K, V, M, O);
}
